// SSCAModule_80985903334103
// MI455X (gfx1250) — matrix operand order repaired and checked by reading, not compiled, not run on hardware
//
#include <hip/hip_runtime.h>

typedef __bf16 bf16;
typedef __attribute__((ext_vector_type(16))) __bf16 v16bf;
typedef __attribute__((ext_vector_type(8)))  float  v8f;

#define DEVINL __device__ __forceinline__

// ---- problem dims ----
#define NB    16
#define HWP   1024      // H*W
#define CIN   2048
#define CPROJ 2048
#define DQ    512
#define NA    64

#define LDL   40        // padded LDS row stride (32 data + 8 pad bf16)

// =====================================================================
// WMMA fragment helpers (v_wmma_f32_16x16x32_bf16, wave32)
// A frag: 16x32, lane<16 holds row M=lane, K chunks {0..7, 16..23};
//         lane>=16 holds row M=lane-16, K chunks {8..15, 24..31}.
// B frag: 32x16 loaded from an N x K row-major source ("Bt"):
//         lane<16 holds col N=lane, K {0..7, 16..23}; lane>=16 holds K {8..15, 24..31}.
// D/C:    lane holds N=lane%16, M = v + 8*(lane/16) for VGPR v.
// =====================================================================

DEVINL v16bf load_a_frag(const bf16* __restrict__ tile, int ld, int lane) {
  const bf16* p = tile + (size_t)(lane & 15) * ld + ((lane & 16) ? 8 : 0);
  union { uint4 q[2]; v16bf v; } u;
  u.q[0] = *(const uint4*)(p);
  u.q[1] = *(const uint4*)(p + 16);
  return u.v;
}

DEVINL v16bf load_b_frag(const bf16* __restrict__ tile, int ld, int lane) {
  const bf16* p = tile + (size_t)(lane & 15) * ld + ((lane & 16) ? 8 : 0);
  union { uint4 q[2]; v16bf v; } u;
  u.q[0] = *(const uint4*)(p);
  u.q[1] = *(const uint4*)(p + 16);
  return u.v;
}

DEVINL v8f wmma_bf16(v16bf a, v16bf b, v8f c) {
  return __builtin_amdgcn_wmma_f32_16x16x32_bf16(false, a, false, b,
                                                 (short)0, c, false, false);
}

DEVINL v8f splat8(float x) {
  v8f r;
#pragma unroll
  for (int v = 0; v < 8; ++v) r[v] = x;
  return r;
}

DEVINL void store_tileT_bf16(bf16* __restrict__ dst, int ld, v8f acc, int lane) {
  bf16* p = dst + (size_t)(lane & 15) * ld + ((lane & 16) ? 8 : 0);
  union { bf16 h[8]; uint4 q; } t;
#pragma unroll
  for (int v = 0; v < 8; ++v) t.h[v] = (bf16)acc[v];
  *(uint4*)p = t.q;
}

DEVINL void store_tileT_f32(float* __restrict__ dst, int ld, v8f acc, int lane) {
  float* p = dst + (size_t)(lane & 15) * ld + ((lane & 16) ? 8 : 0);
  float4 lo = make_float4(acc[0], acc[1], acc[2], acc[3]);
  float4 hi = make_float4(acc[4], acc[5], acc[6], acc[7]);
  *(float4*)(p)     = lo;
  *(float4*)(p + 4) = hi;
}

// =====================================================================
// gfx1250 async global->LDS copy (ASYNCcnt-tracked), GVS addressing:
//   addr = SGPR64 base + VGPR 32-bit byte offset.  VDST VGPR holds the
//   LDS byte address (low 32 bits of the generic __shared__ pointer).
// =====================================================================
DEVINL void async_copy_b128(const bf16* gbase, unsigned gbyte_off, bf16* lptr) {
  unsigned lds_off = (unsigned)(uintptr_t)lptr;
  asm volatile("global_load_async_to_lds_b128 %0, %1, %2 offset:0"
               :
               : "v"(lds_off), "v"(gbyte_off), "s"(gbase)
               : "memory");
}

DEVINL void wait_async_le1() {
  asm volatile("s_wait_asynccnt 0x1" ::: "memory");
}
DEVINL void wait_async_0() {
  asm volatile("s_wait_asynccnt 0x0" ::: "memory");
}

// =====================================================================
// Direct-from-global 32x64 wave tile (used by the small GEMMs).
// =====================================================================
DEVINL void gemm_core_32x64(const bf16* __restrict__ A, int lda,
                            const bf16* __restrict__ Bt, int ldb,
                            int K, int lane, v8f acc[2][4]) {
  for (int k = 0; k < K; k += 32) {
    if (k + 256 < K) {
      __builtin_prefetch(A + k + 256, 0, 1);
      __builtin_prefetch(Bt + k + 256, 0, 1);
    }
    v16bf a0 = load_a_frag(A + k, lda, lane);
    v16bf a1 = load_a_frag(A + (size_t)16 * lda + k, lda, lane);
    v16bf b0 = load_b_frag(Bt + k, ldb, lane);
    v16bf b1 = load_b_frag(Bt + (size_t)16 * ldb + k, ldb, lane);
    v16bf b2 = load_b_frag(Bt + (size_t)32 * ldb + k, ldb, lane);
    v16bf b3 = load_b_frag(Bt + (size_t)48 * ldb + k, ldb, lane);
    acc[0][0] = wmma_bf16(a0, b0, acc[0][0]);
    acc[0][1] = wmma_bf16(a0, b1, acc[0][1]);
    acc[0][2] = wmma_bf16(a0, b2, acc[0][2]);
    acc[0][3] = wmma_bf16(a0, b3, acc[0][3]);
    acc[1][0] = wmma_bf16(a1, b0, acc[1][0]);
    acc[1][1] = wmma_bf16(a1, b1, acc[1][1]);
    acc[1][2] = wmma_bf16(a1, b2, acc[1][2]);
    acc[1][3] = wmma_bf16(a1, b3, acc[1][3]);
  }
}

// =====================================================================
// LDS-staged 32x64 wave tile: the block-uniform Bt tile (64 rows x K)
// is double-buffered through LDS with async copies; each of the 8 waves
// reads B fragments from LDS and its private A rows from global.
// =====================================================================
DEVINL void gemm_core_32x64_ldsB(const bf16* __restrict__ A, int lda,
                                 const bf16* __restrict__ BtG, int ldb,
                                 bf16* smem /* [2][64*LDL] */,
                                 int K, int lane, int tid, v8f acc[2][4]) {
  const int r = tid >> 2;              // 0..63 : Bt row
  const int c = (tid & 3) * 8;         // 0/8/16/24 : bf16 column chunk
  bf16* l0 = smem + r * LDL + c;
  bf16* l1 = smem + 64 * LDL + r * LDL + c;
  const unsigned gstep = (unsigned)((size_t)r * ldb + c) * 2u;

  async_copy_b128(BtG, gstep, l0);     // stage k=0 into buffer 0
  int cur = 0;
  for (int k = 0; k < K; k += 32) {
    bool more = (k + 32 < K);
    if (more)
      async_copy_b128(BtG, gstep + (unsigned)(k + 32) * 2u, cur ? l0 : l1);
    if (more) wait_async_le1(); else wait_async_0();
    __syncthreads();

    const bf16* Bl = smem + cur * (64 * LDL);
    if (k + 256 < K) __builtin_prefetch(A + k + 256, 0, 1);
    v16bf a0 = load_a_frag(A + k, lda, lane);
    v16bf a1 = load_a_frag(A + (size_t)16 * lda + k, lda, lane);
    v16bf b0 = load_b_frag(Bl + 0 * 16 * LDL, LDL, lane);
    v16bf b1 = load_b_frag(Bl + 1 * 16 * LDL, LDL, lane);
    v16bf b2 = load_b_frag(Bl + 2 * 16 * LDL, LDL, lane);
    v16bf b3 = load_b_frag(Bl + 3 * 16 * LDL, LDL, lane);
    acc[0][0] = wmma_bf16(a0, b0, acc[0][0]);
    acc[0][1] = wmma_bf16(a0, b1, acc[0][1]);
    acc[0][2] = wmma_bf16(a0, b2, acc[0][2]);
    acc[0][3] = wmma_bf16(a0, b3, acc[0][3]);
    acc[1][0] = wmma_bf16(a1, b0, acc[1][0]);
    acc[1][1] = wmma_bf16(a1, b1, acc[1][1]);
    acc[1][2] = wmma_bf16(a1, b2, acc[1][2]);
    acc[1][3] = wmma_bf16(a1, b3, acc[1][3]);

    __syncthreads();                   // all reads of 'cur' done before restage
    cur ^= 1;
  }
}

// =====================================================================
// Conversion / transpose kernels
// =====================================================================

__global__ void __launch_bounds__(256)
cvt_kernel(const float* __restrict__ src, bf16* __restrict__ dst, long n) {
  long i = ((long)blockIdx.x * blockDim.x + threadIdx.x) * 4;
  if (i + 4 <= n) {
    float4 f = *(const float4*)(src + i);
    union { bf16 h[4]; uint2 u; } t;
    t.h[0] = (bf16)f.x; t.h[1] = (bf16)f.y; t.h[2] = (bf16)f.z; t.h[3] = (bf16)f.w;
    *(uint2*)(dst + i) = t.u;
  }
}

// src[R][C] f32 -> dst[C][R] bf16
__global__ void __launch_bounds__(256)
transpose_cvt_kernel(const float* __restrict__ src, bf16* __restrict__ dst,
                     int R, int C) {
  int idx = blockIdx.x * blockDim.x + threadIdx.x;
  if (idx < R * C) {
    int c = idx / R, r = idx % R;
    dst[(size_t)c * R + r] = (bf16)src[(size_t)r * C + c];
  }
}

// =====================================================================
// Q projection: Qt[b][a][c] = sum_d queries[b][c][d] * Wq[d][a] + bq[a]
// =====================================================================
__global__ void __launch_bounds__(256)
qgemm_kernel(const bf16* __restrict__ Q, const bf16* __restrict__ WqT,
             const float* __restrict__ bq, bf16* __restrict__ Qt) {
  int gtid = blockIdx.x * 256 + threadIdx.x;
  int wave = gtid >> 5, lane = gtid & 31;
  int mBase = wave * 32;                       // 1024 waves over M=32768
  const bf16* A = Q + (size_t)mBase * DQ;
  v8f acc[2][4];
#pragma unroll
  for (int i = 0; i < 2; ++i)
#pragma unroll
    for (int j = 0; j < 4; ++j) acc[i][j] = splat8(0.0f);
  gemm_core_32x64(A, DQ, WqT, DQ, DQ, lane, acc);

  int b  = mBase / CPROJ;
  int c0 = mBase % CPROJ;
  bf16* Qb = Qt + (size_t)b * NA * CPROJ;
#pragma unroll
  for (int j = 0; j < 4; ++j) {
    v8f bias = splat8(bq[j * 16 + (lane & 15)]);
#pragma unroll
    for (int i = 0; i < 2; ++i) {
      v8f r = acc[i][j] + bias;
      store_tileT_bf16(Qb + (size_t)(j * 16) * CPROJ + c0 + i * 16, CPROJ, r, lane);
    }
  }
}

// =====================================================================
// F conv: F[b][n][cout] = feat[b][n][:] . Wf[:,cout] + bf[cout]
// A = WfT rows (per wave), Bt = features tile (block-uniform, via LDS).
// Block: 8 waves = cout 256-row group; grid = b(16) x ng(16) x mgG(8).
// =====================================================================
__global__ void __launch_bounds__(256)
fgemm_kernel(const bf16* __restrict__ WfT, const bf16* __restrict__ feat,
             const float* __restrict__ bfv, bf16* __restrict__ F) {
  __shared__ bf16 smem[2 * 64 * LDL];
  int tid  = threadIdx.x;
  int wave = tid >> 5, lane = tid & 31;
  int blk  = blockIdx.x;
  int b    = blk >> 7;
  int rem  = blk & 127;
  int ng   = rem >> 3;                         // 16 n groups of 64
  int mg   = (rem & 7) * 8 + wave;             // 64 cout groups of 32
  const bf16* A  = WfT + (size_t)mg * 32 * CIN;
  const bf16* Bt = feat + ((size_t)b * HWP + (size_t)ng * 64) * CIN;
  v8f acc[2][4];
#pragma unroll
  for (int i = 0; i < 2; ++i)
#pragma unroll
    for (int j = 0; j < 4; ++j) acc[i][j] = splat8(0.0f);
  gemm_core_32x64_ldsB(A, CIN, Bt, CIN, smem, CIN, lane, tid, acc);

  int mOff = (lane & 16) ? 8 : 0;
  bf16* Fb = F + (size_t)b * HWP * CPROJ;
#pragma unroll
  for (int i = 0; i < 2; ++i) {
    const float* bp = bfv + mg * 32 + i * 16 + mOff;
    float4 blo = *(const float4*)(bp);
    float4 bhi = *(const float4*)(bp + 4);
    v8f bias;
    bias[0]=blo.x; bias[1]=blo.y; bias[2]=blo.z; bias[3]=blo.w;
    bias[4]=bhi.x; bias[5]=bhi.y; bias[6]=bhi.z; bias[7]=bhi.w;
#pragma unroll
    for (int j = 0; j < 4; ++j) {
      v8f r = acc[i][j] + bias;
      store_tileT_bf16(Fb + (size_t)(ng * 64 + j * 16) * CPROJ + mg * 32 + i * 16,
                       CPROJ, r, lane);
    }
  }
}

// =====================================================================
// S conv, stored transposed: ST[b][cout][n] = feat[b][n][:].Wf1[:,cout]+bf1
// A = features rows (per wave), Bt = Wf1T tile (block-uniform, via LDS).
// Grid = b(16) x ng(32) x mgG(4).
// =====================================================================
__global__ void __launch_bounds__(256)
sgemm_kernel(const bf16* __restrict__ feat, const bf16* __restrict__ Wf1T,
             const float* __restrict__ bf1, bf16* __restrict__ ST) {
  __shared__ bf16 smem[2 * 64 * LDL];
  int tid  = threadIdx.x;
  int wave = tid >> 5, lane = tid & 31;
  int blk  = blockIdx.x;
  int b    = blk >> 7;
  int rem  = blk & 127;
  int ng   = rem >> 2;                         // 32 cout groups of 64
  int mg   = (rem & 3) * 8 + wave;             // 32 n groups of 32
  const bf16* A  = feat + ((size_t)b * HWP + (size_t)mg * 32) * CIN;
  const bf16* Bt = Wf1T + (size_t)ng * 64 * CIN;
  v8f acc[2][4];
#pragma unroll
  for (int i = 0; i < 2; ++i)
#pragma unroll
    for (int j = 0; j < 4; ++j) acc[i][j] = splat8(0.0f);
  gemm_core_32x64_ldsB(A, CIN, Bt, CIN, smem, CIN, lane, tid, acc);

  bf16* Sb = ST + (size_t)b * CPROJ * HWP;
#pragma unroll
  for (int j = 0; j < 4; ++j) {
    v8f bias = splat8(bf1[ng * 64 + j * 16 + (lane & 15)]);
#pragma unroll
    for (int i = 0; i < 2; ++i) {
      v8f r = acc[i][j] + bias;
      store_tileT_bf16(Sb + (size_t)(ng * 64 + j * 16) * HWP + mg * 32 + i * 16,
                       HWP, r, lane);
    }
  }
}

// =====================================================================
// scores + softmax: attn[b][n][a] = softmax_a( (F[b][n][:].Qt[b][a][:])/8 )
// One wave per 16-row strip; full A=64 in registers -> in-wave softmax.
// Also writes attnT[b][a][n] (bf16) for the readout GEMM.
// =====================================================================
__global__ void __launch_bounds__(256)
scores_kernel(const bf16* __restrict__ F, const bf16* __restrict__ Qt,
              float* __restrict__ attn, bf16* __restrict__ attnT) {
  int gtid = blockIdx.x * 256 + threadIdx.x;
  int wave = gtid >> 5, lane = gtid & 31;
  int b = wave >> 6;
  int nBase = (wave & 63) * 16;
  const bf16* Ab = F  + ((size_t)b * HWP + nBase) * CPROJ;
  const bf16* Bt = Qt + (size_t)b * NA * CPROJ;

  v8f acc[4];
#pragma unroll
  for (int j = 0; j < 4; ++j) acc[j] = splat8(0.0f);

  for (int k = 0; k < CPROJ; k += 32) {
    if (k + 256 < CPROJ) __builtin_prefetch(Ab + k + 256, 0, 1);
    v16bf a = load_a_frag(Ab + k, CPROJ, lane);
#pragma unroll
    for (int j = 0; j < 4; ++j) {
      v16bf bq = load_b_frag(Bt + (size_t)(j * 16) * CPROJ + k, CPROJ, lane);
      acc[j] = wmma_bf16(a, bq, acc[j]);
    }
  }

  const float scale = 0.125f;                  // 1/sqrt(64)
#pragma unroll
  for (int v = 0; v < 8; ++v) {
    float t0 = acc[0][v] * scale, t1 = acc[1][v] * scale;
    float t2 = acc[2][v] * scale, t3 = acc[3][v] * scale;
    float mx = fmaxf(fmaxf(t0, t1), fmaxf(t2, t3));
#pragma unroll
    for (int msk = 1; msk <= 8; msk <<= 1)
      mx = fmaxf(mx, __shfl_xor(mx, msk, 32));
    t0 = __expf(t0 - mx); t1 = __expf(t1 - mx);
    t2 = __expf(t2 - mx); t3 = __expf(t3 - mx);
    float s = t0 + t1 + t2 + t3;
#pragma unroll
    for (int msk = 1; msk <= 8; msk <<= 1)
      s += __shfl_xor(s, msk, 32);
    float r = 1.0f / s;
    acc[0][v] = t0 * r; acc[1][v] = t1 * r;
    acc[2][v] = t2 * r; acc[3][v] = t3 * r;
  }

  int mOff = (lane & 16) ? 8 : 0;
#pragma unroll
  for (int j = 0; j < 4; ++j) {
    float* p = attn + ((size_t)(b * HWP + nBase + mOff)) * NA + j * 16 + (lane & 15);
#pragma unroll
    for (int v = 0; v < 8; ++v) p[(size_t)v * NA] = acc[j][v];
    store_tileT_bf16(attnT + ((size_t)b * NA + j * 16) * HWP + nBase,
                     HWP, acc[j], lane);
  }
}

// =====================================================================
// readout: out[b][c][a] = sum_n S[b][n][c] * attn[b][n][a]
// =====================================================================
__global__ void __launch_bounds__(256)
outgemm_kernel(const bf16* __restrict__ attnT, const bf16* __restrict__ ST,
               float* __restrict__ out) {
  int gtid = blockIdx.x * 256 + threadIdx.x;
  int wave = gtid >> 5, lane = gtid & 31;
  int b = wave >> 6;
  int rem = wave & 63;
  int mi = rem >> 5;
  int ng = rem & 31;
  const bf16* A  = attnT + ((size_t)b * NA + (size_t)mi * 32) * HWP;
  const bf16* Bt = ST + ((size_t)b * CPROJ + (size_t)ng * 64) * HWP;
  v8f acc[2][4];
#pragma unroll
  for (int i = 0; i < 2; ++i)
#pragma unroll
    for (int j = 0; j < 4; ++j) acc[i][j] = splat8(0.0f);
  gemm_core_32x64(A, HWP, Bt, HWP, HWP, lane, acc);

  float* ob = out + (size_t)b * CPROJ * NA;
#pragma unroll
  for (int j = 0; j < 4; ++j)
#pragma unroll
    for (int i = 0; i < 2; ++i)
      store_tileT_f32(ob + (size_t)(ng * 64 + j * 16) * NA + mi * 32 + i * 16,
                      NA, acc[i][j], lane);
}

// =====================================================================
// launch
// =====================================================================
extern "C" void kernel_launch(void* const* d_in, const int* in_sizes, int n_in,
                              void* d_out, int out_size, void* d_ws, size_t ws_size,
                              hipStream_t stream) {
  const float* queries  = (const float*)d_in[0];   // [16,2048,512]
  const float* features = (const float*)d_in[1];   // [16,32,32,2048]
  const float* Wq  = (const float*)d_in[2];        // [512,64]
  const float* bq  = (const float*)d_in[3];        // [64]
  const float* Wf  = (const float*)d_in[4];        // [2048,2048]
  const float* bfv = (const float*)d_in[5];        // [2048]
  const float* Wf1 = (const float*)d_in[6];        // [2048,2048]
  const float* bf1 = (const float*)d_in[7];        // [2048]

  float* attn = (float*)d_out;                             // [16,32,32,64]
  float* outp = attn + (size_t)NB * HWP * NA;              // [16,2048,64]

  char* ws = (char*)d_ws;
  bf16* feat_bf = (bf16*)ws; ws += (size_t)NB * HWP * CIN * 2;
  bf16* qry_bf  = (bf16*)ws; ws += (size_t)NB * CPROJ * DQ * 2;
  bf16* WfT_    = (bf16*)ws; ws += (size_t)CIN * CPROJ * 2;
  bf16* Wf1T_   = (bf16*)ws; ws += (size_t)CIN * CPROJ * 2;
  bf16* WqT_    = (bf16*)ws; ws += (size_t)DQ * NA * 2;
  bf16* Qt      = (bf16*)ws; ws += (size_t)NB * NA * CPROJ * 2;
  bf16* Fbuf    = (bf16*)ws; ws += (size_t)NB * HWP * CPROJ * 2;
  bf16* STbuf   = (bf16*)ws; ws += (size_t)NB * CPROJ * HWP * 2;
  bf16* attnT   = (bf16*)ws; ws += (size_t)NB * NA * HWP * 2;

  const long n_feat = (long)NB * HWP * CIN;     // 33,554,432
  const long n_qry  = (long)NB * CPROJ * DQ;    // 16,777,216

  cvt_kernel<<<(int)(n_feat / 1024), 256, 0, stream>>>(features, feat_bf, n_feat);
  cvt_kernel<<<(int)(n_qry  / 1024), 256, 0, stream>>>(queries,  qry_bf,  n_qry);
  transpose_cvt_kernel<<<(CIN * CPROJ + 255) / 256, 256, 0, stream>>>(Wf,  WfT_,  CIN, CPROJ);
  transpose_cvt_kernel<<<(CIN * CPROJ + 255) / 256, 256, 0, stream>>>(Wf1, Wf1T_, CIN, CPROJ);
  transpose_cvt_kernel<<<(DQ * NA + 255) / 256, 256, 0, stream>>>(Wq, WqT_, DQ, NA);

  qgemm_kernel<<<128,  256, 0, stream>>>(qry_bf, WqT_, bq, Qt);
  fgemm_kernel<<<2048, 256, 0, stream>>>(WfT_, feat_bf, bfv, Fbuf);
  sgemm_kernel<<<2048, 256, 0, stream>>>(feat_bf, Wf1T_, bf1, STbuf);
  scores_kernel<<<128, 256, 0, stream>>>(Fbuf, Qt, attn, attnT);
  outgemm_kernel<<<128, 256, 0, stream>>>(attnT, STbuf, outp);
}
